// BilinearDecoder_89026082111807
// MI455X (gfx1250) — matrix operand order repaired and checked by reading, not compiled, not run on hardware
//
#include <hip/hip_runtime.h>
#include <hip/hip_bf16.h>

// Static problem dims (match reference)
#define NG    4096   // graphs
#define NPG   64     // nodes per graph
#define KC    32     // candidates per graph
#define NF    128    // feature dim
#define DIN_  32
#define DOUT_ 32
#define GEMB_ 64
#define DHID_ 64
#define HCAT  (DOUT_ + GEMB_)   // 96

typedef __attribute__((ext_vector_type(16))) _Float16 v16h;
typedef __attribute__((ext_vector_type(8)))  float    v8f;

static __device__ __forceinline__ v8f wmma_f16(v16h a, v16h b, v8f c) {
  // D = A(16x32 f16) x B(32x16 f16) + C(16x16 f32)
  return __builtin_amdgcn_wmma_f32_16x16x32_f16(false, a, false, b, (short)0, c,
                                                false, false);
}

// A-matrix fragment from row-major f16 LDS tile.
// Lane l: row = 16*mt + (l&15); elements i = 0..15 hold K = kb + 16*lh + i.
static __device__ __forceinline__ v16h a_frag(const _Float16* base, int stride,
                                              int mt, int kb, int lane) {
  const int row = mt * 16 + (lane & 15);
  const int lh  = lane >> 4;
  const _Float16* p = base + row * stride + kb + 16 * lh;
  v16h a;
#pragma unroll
  for (int i = 0; i < 8; ++i) a[i] = p[i];
#pragma unroll
  for (int i = 0; i < 8; ++i) a[8 + i] = p[8 + i];
  return a;
}

// B-matrix fragment from row-major f16 LDS tile [K x N].
// half i: K = kb + i + 16*lh, N = 16*nt + (lane&15).
static __device__ __forceinline__ v16h b_frag_lds(const _Float16* base, int stride,
                                                  int nt, int kb, int lane) {
  const int col = nt * 16 + (lane & 15);
  const int k0  = kb + 16 * (lane >> 4);
  v16h b;
#pragma unroll
  for (int i = 0; i < 16; ++i) b[i] = base[(k0 + i) * stride + col];
  return b;
}

// B-matrix fragment straight from a row-major fp32 global weight [K x N].
static __device__ __forceinline__ v16h b_frag_gf32(const float* base, int stride,
                                                   int nt, int kb, int lane) {
  const int col = nt * 16 + (lane & 15);
  const int k0  = kb + 16 * (lane >> 4);
  v16h b;
#pragma unroll
  for (int i = 0; i < 16; ++i) b[i] = (_Float16)base[(k0 + i) * stride + col];
  return b;
}

static __device__ __forceinline__ v8f zero8() {
  v8f c = {0.f, 0.f, 0.f, 0.f, 0.f, 0.f, 0.f, 0.f};
  return c;
}

// vbuf is dead after the layer-1 GEMM (two barriers before hbuf is written),
// so the two largest tiles share one 8 KB LDS region.
union StageUnion {
  _Float16 vbuf[KC][NF];   // gathered x_t rows (f16)        8 KB
  _Float16 hbuf[KC][HCAT]; // [sim | g_emb] decoder input    6 KB
};

__global__ __launch_bounds__(32)
void BilinearDecoder_89026082111807_kernel(
    const float* __restrict__ x_q,  const float* __restrict__ x_t,
    const float* __restrict__ g_emb,
    const float* __restrict__ W1,   const float* __restrict__ b1,
    const float* __restrict__ W2,   const float* __restrict__ b2,
    const float* __restrict__ M,    // bilinear_mat [32,32,32]
    const float* __restrict__ Wd1,  const float* __restrict__ bd1,
    const float* __restrict__ Wd2,  const float* __restrict__ bd2,
    const int*   __restrict__ u_local, const int* __restrict__ v_local,
    float* __restrict__ out) {
  const int g    = blockIdx.x;
  const int lane = threadIdx.x;  // wave32, one wave per block

  __shared__ StageUnion uni;                  // 8 KB (vbuf / hbuf)
  __shared__ _Float16 h1buf[KC][DIN_];        // relu(v@W1+b1)   2 KB
  __shared__ _Float16 ztbuf[KC][DIN_];        // zt              2 KB
  __shared__ _Float16 mqbuf[DIN_][DOUT_];     // Mq[l][j]        2 KB
  __shared__ float    ubuf[NF];
  __shared__ float    h1q[DIN_];
  __shared__ float    zqv[DIN_];
  __shared__ float    outacc[KC];             // final reduction (ds_add_f32)

  // Warm L2 for the shared bilinear tensor (128 KB, reused by all 4096 blocks).
  __builtin_prefetch(M + (size_t)lane * 1024, 0, 3);

  // ---- Stage gathers: 32 candidate rows of x_t (row = lane) + the u row ----
  {
    const int vg = v_local[g * KC + lane];
    const float4* vp =
        reinterpret_cast<const float4*>(x_t + (size_t)(g * NPG + vg) * NF);
#pragma unroll 8
    for (int f4 = 0; f4 < NF / 4; ++f4) {
      float4 t = vp[f4];
      uni.vbuf[lane][f4 * 4 + 0] = (_Float16)t.x;
      uni.vbuf[lane][f4 * 4 + 1] = (_Float16)t.y;
      uni.vbuf[lane][f4 * 4 + 2] = (_Float16)t.z;
      uni.vbuf[lane][f4 * 4 + 3] = (_Float16)t.w;
    }
    const int ug = u_local[g];
    const float* up = x_q + (size_t)(g * NPG + ug) * NF;
    ubuf[lane]      = up[lane];
    ubuf[32 + lane] = up[32 + lane];
    ubuf[64 + lane] = up[64 + lane];
    ubuf[96 + lane] = up[96 + lane];
    outacc[lane] = 0.f;
  }
  __syncthreads();

  // ---- zq = MLP(u_emb): single row -> VALU matvec (lane = out channel) ----
  {
    float acc = b1[lane];
#pragma unroll 8
    for (int f = 0; f < NF; ++f) acc = fmaf(ubuf[f], W1[f * DIN_ + lane], acc);
    h1q[lane] = fmaxf(acc, 0.f);
  }
  __syncthreads();
  {
    float acc = b2[lane];
#pragma unroll
    for (int d = 0; d < DIN_; ++d) acc = fmaf(h1q[d], W2[d * DIN_ + lane], acc);
    zqv[lane] = acc;
  }

  // ---- zt layer 1: [32,128] @ W1[128,32], WMMA f16 -> f32 ----
  {
#pragma unroll
    for (int nt = 0; nt < 2; ++nt) {
#pragma unroll
      for (int mt = 0; mt < 2; ++mt) {
        v8f c = zero8();
#pragma unroll
        for (int kc = 0; kc < 4; ++kc)
          c = wmma_f16(a_frag(&uni.vbuf[0][0], NF, mt, kc * 32, lane),
                       b_frag_gf32(W1, DIN_, nt, kc * 32, lane), c);
        const int col   = nt * 16 + (lane & 15);
        const int rbase = mt * 16 + 8 * (lane >> 4);
        const float bias = b1[col];
#pragma unroll
        for (int r = 0; r < 8; ++r)
          h1buf[rbase + r][col] = (_Float16)fmaxf(c[r] + bias, 0.f);
      }
    }
  }
  __syncthreads();  // h1buf + zqv visible; vbuf now dead

  // ---- zt layer 2: [32,32] @ W2[32,32] ----
  {
#pragma unroll
    for (int nt = 0; nt < 2; ++nt) {
#pragma unroll
      for (int mt = 0; mt < 2; ++mt) {
        v8f c = wmma_f16(a_frag(&h1buf[0][0], DIN_, mt, 0, lane),
                         b_frag_gf32(W2, DIN_, nt, 0, lane), zero8());
        const int col   = nt * 16 + (lane & 15);
        const int rbase = mt * 16 + 8 * (lane >> 4);
        const float bias = b2[col];
#pragma unroll
        for (int r = 0; r < 8; ++r)
          ztbuf[rbase + r][col] = (_Float16)(c[r] + bias);
      }
    }
  }

  // ---- Mq[l][j] = sum_k zq[k] * M[k,l,j]  (3-tensor x vector: VALU, L2-hot M)
  {
    const float* Mj = M + lane;
#pragma unroll 2
    for (int l = 0; l < DIN_; ++l) {
      float acc = 0.f;
#pragma unroll
      for (int k = 0; k < DIN_; ++k)
        acc = fmaf(zqv[k], Mj[k * (DIN_ * DOUT_) + l * DOUT_], acc);
      mqbuf[l][lane] = (_Float16)acc;
    }
  }
  __syncthreads();

  // ---- sim = zt @ Mq : [32,32]@[32,32] -> hbuf[:, 0:32] ----
  {
#pragma unroll
    for (int nt = 0; nt < 2; ++nt) {
      v16h bfr = b_frag_lds(&mqbuf[0][0], DOUT_, nt, 0, lane);
#pragma unroll
      for (int mt = 0; mt < 2; ++mt) {
        v8f c = wmma_f16(a_frag(&ztbuf[0][0], DIN_, mt, 0, lane), bfr, zero8());
        const int col   = nt * 16 + (lane & 15);
        const int rbase = mt * 16 + 8 * (lane >> 4);
#pragma unroll
        for (int r = 0; r < 8; ++r) uni.hbuf[rbase + r][col] = (_Float16)c[r];
      }
    }
    // broadcast graph embedding into hbuf[:, 32:96]
    const _Float16 ge0 = (_Float16)g_emb[g * GEMB_ + lane];
    const _Float16 ge1 = (_Float16)g_emb[g * GEMB_ + 32 + lane];
#pragma unroll 8
    for (int r = 0; r < KC; ++r) {
      uni.hbuf[r][DOUT_ + lane]      = ge0;
      uni.hbuf[r][DOUT_ + 32 + lane] = ge1;
    }
  }
  __syncthreads();

  // ---- decoder: relu([32,96] @ Wd1 + bd1) dotted with Wd2 on the fly.
  // Each lane owns one hidden column; its relu(hd)*Wd2[col] partials are
  // reduced across the 64 hidden dims straight into outacc via ds_add_f32.
  {
#pragma unroll
    for (int nt = 0; nt < 4; ++nt) {
      const int col    = nt * 16 + (lane & 15);
      const float bias = bd1[col];
      const float w2v  = Wd2[col];
#pragma unroll
      for (int mt = 0; mt < 2; ++mt) {
        v8f c = zero8();
#pragma unroll
        for (int kc = 0; kc < 3; ++kc)
          c = wmma_f16(a_frag(&uni.hbuf[0][0], HCAT, mt, kc * 32, lane),
                       b_frag_gf32(Wd1, DHID_, nt, kc * 32, lane), c);
        const int rbase = mt * 16 + 8 * (lane >> 4);
#pragma unroll
        for (int r = 0; r < 8; ++r)
          atomicAdd(&outacc[rbase + r], fmaxf(c[r] + bias, 0.f) * w2v);
      }
    }
  }
  __syncthreads();

  // ---- final: out[p] = outacc[row] + bd2 ----
  out[(size_t)g * KC + lane] = outacc[lane] + bd2[0];
}

extern "C" void kernel_launch(void* const* d_in, const int* in_sizes, int n_in,
                              void* d_out, int out_size, void* d_ws, size_t ws_size,
                              hipStream_t stream) {
  (void)in_sizes; (void)n_in; (void)out_size; (void)d_ws; (void)ws_size;
  const float* x_q     = (const float*)d_in[0];
  const float* x_t     = (const float*)d_in[1];
  const float* g_emb   = (const float*)d_in[2];
  const float* W1      = (const float*)d_in[3];
  const float* b1      = (const float*)d_in[4];
  const float* W2      = (const float*)d_in[5];
  const float* b2      = (const float*)d_in[6];
  const float* M       = (const float*)d_in[7];
  const float* Wd1     = (const float*)d_in[8];
  const float* bd1     = (const float*)d_in[9];
  const float* Wd2     = (const float*)d_in[10];
  const float* bd2     = (const float*)d_in[11];
  const int*   u_local = (const int*)d_in[12];
  const int*   v_local = (const int*)d_in[13];

  BilinearDecoder_89026082111807_kernel<<<NG, 32, 0, stream>>>(
      x_q, x_t, g_emb, W1, b1, W2, b2, M, Wd1, bd1, Wd2, bd2,
      u_local, v_local, (float*)d_out);
}
